// LabelWiseContrastiveLoss_37752762532272
// MI455X (gfx1250) — matrix operand order repaired and checked by reading, not compiled, not run on hardware
//
#include <hip/hip_runtime.h>
#include <hip/hip_bf16.h>

// CDNA5 / gfx1250 — wave32, WMMA 16x16x32 bf16 (f32 acc), TDM async B-panel staging.

typedef __attribute__((ext_vector_type(16))) __bf16 v16bf;
typedef __attribute__((ext_vector_type(8)))  __bf16 v8bf;
typedef __attribute__((ext_vector_type(8)))  float  v8f;
typedef __attribute__((ext_vector_type(4)))  unsigned v4u;
typedef __attribute__((ext_vector_type(8)))  int    v8i;
typedef __attribute__((ext_vector_type(4)))  int    v4i;

#define TAU_F 0.1f
#define KC    128      // K-chunk staged per TDM op (4 WMMA steps)

// ---------- device helpers ----------

__device__ __forceinline__ unsigned fkey(float f) {
    unsigned u = __float_as_uint(f);
    return (u & 0x80000000u) ? ~u : (u | 0x80000000u);
}
__device__ __forceinline__ float keyf(unsigned k) {
    return __uint_as_float((k & 0x80000000u) ? (k & 0x7FFFFFFFu) : ~k);
}
__device__ __forceinline__ __bf16 f2bf(float x) {
    unsigned u = __float_as_uint(x);
    unsigned r = u + 0x7FFFu + ((u >> 16) & 1u);
    unsigned short h = (unsigned short)(r >> 16);
    return __builtin_bit_cast(__bf16, h);
}
__device__ __forceinline__ float wsum(float v) {
    #pragma unroll
    for (int m = 16; m; m >>= 1) v += __shfl_xor(v, m, 32);
    return v;
}
__device__ __forceinline__ int wsumi(int v) {
    #pragma unroll
    for (int m = 16; m; m >>= 1) v += __shfl_xor(v, m, 32);
    return v;
}
__device__ __forceinline__ float wmaxf(float v) {
    #pragma unroll
    for (int m = 16; m; m >>= 1) v = fmaxf(v, __shfl_xor(v, m, 32));
    return v;
}

// Tensor Data Mover: async load of a (rows x 16)-element bf16 tile (row stride = C elems)
// from global pT into LDS at lds_off. 2-D D# descriptor, groups 2/3 zero.
__device__ __forceinline__ void tdm_load_tile(const __bf16* gsrc, unsigned lds_off,
                                              int C, int D, int rows) {
    unsigned long long ga = (unsigned long long)gsrc;
    v4u g0;
    g0[0] = 1u;                       // count=1, is_restore=0, gather=0
    g0[1] = lds_off;                  // LDS byte address (wave-relative)
    g0[2] = (unsigned)ga;             // global_addr[31:0]
    g0[3] = (unsigned)((ga >> 32) & 0x01FFFFFFull) | 0x80000000u; // addr[56:32] | type=2
    v8i g1;
    g1[0] = (int)(1u << 16);                                   // data_size=1 (2 bytes)
    g1[1] = (int)(((unsigned)C & 0xFFFFu) << 16);              // tensor_dim0 lo16
    g1[2] = (int)((((unsigned)C) >> 16) | (((unsigned)D & 0xFFFFu) << 16)); // dim0 hi / dim1 lo
    g1[3] = (int)((((unsigned)D) >> 16) | (16u << 16));        // dim1 hi | tile_dim0=16
    g1[4] = (int)((unsigned)rows & 0xFFFFu);                   // tile_dim1 | tile_dim2=0
    g1[5] = (int)(unsigned)C;                                  // tensor_dim0_stride lo32
    g1[6] = 0;                                                 // stride hi | dim1_stride lo
    g1[7] = 0;
    v4i gz = {0, 0, 0, 0};
#if __clang_major__ >= 23
    v8i gz8 = {0, 0, 0, 0, 0, 0, 0, 0};
    __builtin_amdgcn_tensor_load_to_lds(g0, g1, gz, gz, gz8, 0);
#else
    __builtin_amdgcn_tensor_load_to_lds(g0, g1, gz, gz, 0);
#endif
}

// ---------- kernel 1: normalize label_proto, store transposed bf16 pT[D][C] ----------

__global__ void k_protoprep(const float* __restrict__ proto, __bf16* __restrict__ pT,
                            int C, int D) {
    const int row = blockIdx.x;
    const int tid = threadIdx.x;    // 128
    __shared__ float red[128];
    float s = 0.f;
    for (int k = tid; k < D; k += 128) { float v = proto[(size_t)row * D + k]; s += v * v; }
    red[tid] = s;
    __syncthreads();
    for (int off = 64; off; off >>= 1) {
        if (tid < off) red[tid] += red[tid + off];
        __syncthreads();
    }
    const float inv = 1.0f / fmaxf(sqrtf(red[0]), 1e-12f);
    for (int k = tid; k < D; k += 128)
        pT[(size_t)k * C + row] = f2bf(proto[(size_t)row * D + k] * inv);
}

// ---------- kernel 2: per-batch random-negative membership (top-Kr of masked rand scores) ----------

__global__ void k_randsel(const int* __restrict__ targets, const float* __restrict__ rsc,
                          unsigned char* __restrict__ rmember, int C, int Kr) {
    const int b = blockIdx.x;
    const int lane = threadIdx.x;   // 1 wave
    const int T = C / 32;
    const int*   tg = targets + (size_t)b * C;
    const float* rs = rsc + (size_t)b * C;
    unsigned char* rm = rmember + (size_t)b * C;

    int nmask = 0;
    for (int t = 0; t < T; ++t) { int j = lane + t * 32; nmask += (tg[j] == 0); }
    nmask = wsumi(nmask);
    const int Ke = Kr < nmask ? Kr : nmask;

    unsigned lo = 0, hi = 0xFFFFFFFFu;
    if (Ke > 0) {
        for (int it = 0; it < 32 && lo < hi; ++it) {
            unsigned d = hi - lo;
            unsigned mid = lo + (d >> 1) + (d & 1u);
            int c = 0;
            for (int t = 0; t < T; ++t) {
                int j = lane + t * 32;
                if (tg[j] == 0) c += (fkey(rs[j]) >= mid);
            }
            c = wsumi(c);
            if (c >= Ke) lo = mid; else hi = mid - 1;
        }
    }
    const unsigned kth = lo;
    int cgt = 0;
    for (int t = 0; t < T; ++t) {
        int j = lane + t * 32;
        bool m = (tg[j] == 0);
        bool sel = m && (Ke > 0) && (fkey(rs[j]) > kth);
        rm[j] = sel ? (unsigned char)1 : (unsigned char)0;
        cgt += sel ? 1 : 0;
    }
    cgt = wsumi(cgt);
    if (lane == 0 && Ke > 0) {
        int need = Ke - cgt;   // ties at kth value in index order (matches top_k)
        for (int j = 0; j < C && need > 0; ++j)
            if (tg[j] == 0 && fkey(rs[j]) == kth) { rm[j] = 1; --need; }
    }
}

// ---------- kernel 3: fused normalize + TDM-staged WMMA sim row-block + exact top-k LSE ----------

__global__ void k_main(const float* __restrict__ feat, const __bf16* __restrict__ pT,
                       const int* __restrict__ targets, const unsigned char* __restrict__ rmember,
                       float* __restrict__ per_anchor, int B, int C, int D, int Kh) {
    const int tilesPerB = C / 16;
    const int b  = blockIdx.x / tilesPerB;
    const int c0 = (blockIdx.x % tilesPerB) * 16;
    const int tid  = threadIdx.x;   // 128
    const int lane = tid & 31;
    const int wave = tid >> 5;      // 4 waves

    extern __shared__ char smem[];
    float*         simrow = (float*)smem;                      // 16*C f32
    float*         nred   = simrow + 16 * C;                   // 128
    float*         inv16  = nred + 128;                        // 16
    __bf16*        At     = (__bf16*)(inv16 + 16);             // 16*D bf16
    __bf16*        Bb     = At + 16 * D;                       // 4 waves * 2 * (KC*16) bf16
    unsigned char* nm     = (unsigned char*)(Bb + 4 * 2 * KC * 16); // C
    unsigned char* rm     = nm + C;                            // C

    // masks into LDS
    for (int j = tid; j < C; j += 128) {
        nm[j] = (targets[(size_t)b * C + j] == 0) ? 1 : 0;
        rm[j] = rmember[(size_t)b * C + j];
    }

    // normalize 16 anchor rows -> bf16 A tile (8 threads per row)
    const int m    = tid >> 3;
    const int part = tid & 7;
    const int seg  = D / 8;
    const float* frow = feat + ((size_t)b * C + c0 + m) * D;
    {
        float s = 0.f;
        for (int i = 0; i < seg; ++i) { float v = frow[part * seg + i]; s += v * v; }
        nred[tid] = s;
    }
    __syncthreads();
    if (tid < 16) {
        float t = 0.f;
        for (int i = 0; i < 8; ++i) t += nred[tid * 8 + i];
        inv16[tid] = 1.0f / fmaxf(sqrtf(t), 1e-12f);
    }
    __syncthreads();
    {
        const float inv = inv16[m];
        for (int i = 0; i < seg; ++i)
            At[m * D + part * seg + i] = f2bf(frow[part * seg + i] * inv);
    }
    __syncthreads();

    // --- WMMA sweep: wave w owns n-tiles w, w+4, ...; B panel staged by TDM, double-buffered ---
    const int nchunks = D / KC;                 // K chunks of 128 (4 WMMA steps each)
    const int arow  = lane & 15;
    const int koff  = (lane < 16) ? 0 : 8;      // A frag K-half selection
    const int mbase = (lane < 16) ? 0 : 8;      // D frag M base
    __bf16* mybuf = Bb + wave * (2 * KC * 16);
    const unsigned mybuf_lds0 = (unsigned)(unsigned long long)(void*)mybuf;
    const unsigned mybuf_lds1 = (unsigned)(unsigned long long)(void*)(mybuf + KC * 16);

    for (int nt = wave; nt < C / 16; nt += 4) {
        const __bf16* gpanel = pT + (size_t)nt * 16;   // column block start
        // prologue: stage chunk 0
        tdm_load_tile(gpanel, mybuf_lds0, C, D, KC);
        v8f acc = {0.f, 0.f, 0.f, 0.f, 0.f, 0.f, 0.f, 0.f};
        const __bf16* arowp = At + arow * D + koff;
        for (int cc = 0; cc < nchunks; ++cc) {
            if (cc + 1 < nchunks) {
                tdm_load_tile(gpanel + (size_t)(cc + 1) * KC * C,
                              ((cc + 1) & 1) ? mybuf_lds1 : mybuf_lds0, C, D, KC);
                __builtin_amdgcn_s_wait_tensorcnt(1);  // chunk cc is resident
            } else {
                __builtin_amdgcn_s_wait_tensorcnt(0);
            }
            const __bf16* bp = mybuf + (cc & 1) * (KC * 16);
            #pragma unroll
            for (int k4 = 0; k4 < KC / 32; ++k4) {
                const int ks = cc * (KC / 32) + k4;
                union { v16bf v; v8bf h[2]; } a;
                a.h[0] = *(const v8bf*)(arowp + ks * 32);
                a.h[1] = *(const v8bf*)(arowp + ks * 32 + 16);
                v16bf bf;
                #pragma unroll
                for (int s = 0; s < 16; ++s)
                    bf[s] = bp[(k4 * 32 + koff + (s & 7) + ((s >> 3) << 4)) * 16 + arow];
                acc = __builtin_amdgcn_wmma_f32_16x16x32_bf16(
                    false, a.v, false, bf, (short)0, acc, false, false);
            }
        }
        const int col = nt * 16 + (lane & 15);
        #pragma unroll
        for (int j = 0; j < 8; ++j)
            simrow[(size_t)(mbase + j) * C + col] = acc[j] * (1.0f / TAU_F);
    }
    __syncthreads();

    // --- exact selection + LSE per positive anchor (wave w handles rows w, w+4, ...) ---
    const int T = C / 32;
    for (int r = wave; r < 16; r += 4) {
        const int c = c0 + r;
        float per = 0.0f;
        if (targets[(size_t)b * C + c] != 0) {
            const float* sr = simrow + (size_t)r * C;
            float mx = -3.4e38f;
            int nmsk = 0;
            for (int t = 0; t < T; ++t) {
                int j = lane + t * 32;
                if (nm[j]) { mx = fmaxf(mx, sr[j]); ++nmsk; }
            }
            mx = wmaxf(mx);
            nmsk = wsumi(nmsk);
            const int Ke = Kh < nmsk ? Kh : nmsk;

            unsigned lo = 0, hi = 0xFFFFFFFFu;
            if (Ke > 0) {
                for (int it = 0; it < 32 && lo < hi; ++it) {
                    unsigned d = hi - lo;
                    unsigned mid = lo + (d >> 1) + (d & 1u);
                    int cge = 0;
                    for (int t = 0; t < T; ++t) {
                        int j = lane + t * 32;
                        if (nm[j]) cge += (fkey(sr[j]) >= mid);
                    }
                    cge = wsumi(cge);
                    if (cge >= Ke) lo = mid; else hi = mid - 1;
                }
            }
            const unsigned kth = lo;
            const float skth = keyf(kth);

            float hs = 0.f, rsum = 0.f;
            int cgt = 0;
            for (int t = 0; t < T; ++t) {
                int j = lane + t * 32;
                if (nm[j]) {
                    float v = sr[j];
                    if (Ke > 0 && fkey(v) > kth) { hs += expf(v - mx); ++cgt; }
                    if (rm[j]) rsum += expf(v - mx);   // hard/rand duplicates counted twice (ref concat)
                }
            }
            hs = wsum(hs);
            rsum = wsum(rsum);
            cgt = wsumi(cgt);
            if (Ke > 0) hs += (float)(Ke - cgt) * expf(skth - mx);

            const float lse = mx + logf(hs + rsum);
            const float pos = sr[c];
            const float d = lse - pos;   // logaddexp(pos,lse) - pos, stable
            per = fmaxf(d, 0.f) + log1pf(expf(-fabsf(d)));
        }
        if (lane == 0) per_anchor[(size_t)b * C + c] = per;
    }
}

// ---------- kernel 4: deterministic reduction to scalar ----------

__global__ void k_final(const float* __restrict__ per_anchor, const int* __restrict__ targets,
                        float* __restrict__ out, int B, int C) {
    __shared__ float rs[256];
    __shared__ float rc[256];
    const int tid = threadIdx.x;
    float total = 0.f;
    for (int b = 0; b < B; ++b) {
        float s = 0.f, cnt = 0.f;
        for (int j = tid; j < C; j += 256) {
            s += per_anchor[(size_t)b * C + j];
            cnt += (targets[(size_t)b * C + j] != 0) ? 1.f : 0.f;
        }
        rs[tid] = s; rc[tid] = cnt;
        __syncthreads();
        for (int o = 128; o; o >>= 1) {
            if (tid < o) { rs[tid] += rs[tid + o]; rc[tid] += rc[tid + o]; }
            __syncthreads();
        }
        if (tid == 0) total += rs[0] / fmaxf(rc[0], 1.f);
        __syncthreads();
    }
    if (tid == 0) out[0] = total / (float)B;
}

// ---------- host launcher ----------

extern "C" void kernel_launch(void* const* d_in, const int* in_sizes, int n_in,
                              void* d_out, int out_size, void* d_ws, size_t ws_size,
                              hipStream_t stream) {
    const float* feat    = (const float*)d_in[0];  // (B,C,D) f32
    const float* proto   = (const float*)d_in[1];  // (C,D)   f32
    const int*   targets = (const int*)d_in[2];    // (B,C)   i32
    const float* rsc     = (const float*)d_in[3];  // (B,C)   f32
    float* out = (float*)d_out;

    const int CD = in_sizes[1];
    const int B  = in_sizes[0] / CD;
    const int C  = in_sizes[2] / B;
    const int D  = CD / C;
    const int K  = C - 1;
    const int Kh = (int)((double)K * 0.3);   // 306 for C=1024
    const int Kr = K - Kh;                   // 717

    char* ws = (char*)d_ws;
    __bf16* pT = (__bf16*)ws;
    size_t pTb = (size_t)C * D * sizeof(__bf16);
    unsigned char* rmem = (unsigned char*)(ws + pTb);
    size_t rb = ((size_t)B * C + 255) & ~(size_t)255;
    float* per = (float*)(ws + pTb + rb);

    k_protoprep<<<C, 128, 0, stream>>>(proto, pT, C, D);
    k_randsel<<<B, 32, 0, stream>>>(targets, rsc, rmem, C, Kr);

    const size_t smem = (size_t)16 * C * sizeof(float)         // sim row-block
                      + (size_t)144 * sizeof(float)            // nred + inv16
                      + (size_t)16 * D * sizeof(__bf16)        // bf16 A tile
                      + (size_t)4 * 2 * KC * 16 * sizeof(__bf16) // TDM B double-buffers
                      + (size_t)2 * C;                         // masks
    (void)hipFuncSetAttribute(reinterpret_cast<const void*>(k_main),
                              hipFuncAttributeMaxDynamicSharedMemorySize, (int)smem);
    k_main<<<B * (C / 16), 128, smem, stream>>>(feat, pT, targets, rmem, per, B, C, D, Kh);

    k_final<<<1, 256, 0, stream>>>(per, targets, out, B, C);
}
